// GATLayer_35141422416077
// MI455X (gfx1250) — matrix operand order repaired and checked by reading, not compiled, not run on hardware
//
#include <hip/hip_runtime.h>

// ---------------------------------------------------------------------------
// GAT layer, MI455X (gfx1250, wave32, WMMA 16x16x32 f16).
//   h:(4096,512)f32  adj:(4096,4096)i32  W:(512,512)f32  a_src/a_dst:(4,128)f32
//   out:(4096,512)f32
// Pipeline:
//   cvt          : h,W -> f16
//   gemm_hprime  : hp = h @ W.T (WMMA), also hpT (f16, K-major for attn GEMM B)
//   attn_logits  : s[i,h], d[i,h] dot products, pre-scaled by log2(e)
//   softmax_stats: two-phase masked softmax stats (fmax pass, then 1 exp/elem)
//   attn_gemm    : out = ELU( softmax(mask(leaky(s_i+d_j))) @ h_prime ), WMMA,
//                  2-way j-split per tile with LDS combine.
// All score math lives in the log2 domain: p = exp2(leaky2(s2+d2) - m2) * 1/l,
// valid because leaky/masking commute with multiplication by log2(e) > 0.
// ---------------------------------------------------------------------------

#define N_NODES  4096
#define IN_DIM   512
#define OUT_DIM  512
#define NHEADS   4
#define HEAD_DIM 128
#define ALPHA    0.2f
#define NEG_INF  (-9.0e15f)
#define LOG2E    1.4426950408889634f

typedef __attribute__((ext_vector_type(16))) _Float16 v16h;
typedef __attribute__((ext_vector_type(8)))  _Float16 v8h;
typedef __attribute__((ext_vector_type(8)))  float    v8f;
typedef __attribute__((ext_vector_type(4)))  int      v4i;
typedef __attribute__((ext_vector_type(4)))  float    v4f;

// ---------------------------------------------------------------- f32 -> f16
__global__ __launch_bounds__(256) void cvt_f32_f16(const float* __restrict__ src,
                                                   _Float16* __restrict__ dst, int n) {
  int i = blockIdx.x * blockDim.x + threadIdx.x;
  if (i < n) dst[i] = (_Float16)src[i];
}

// ------------------------------------------------- hp = h @ W.T  (WMMA f16)
// One wave per 16x16 C tile. A lane mapping (16-bit A 16x32):
//   lane L<16 : row L,   k slots {0..7, 16..23}
//   lane L>=16: row L-16,k slots {8..15,24..31}
// B lane mapping (32x16): lane = column, k slots as for A (hi lanes +8)
// => both operands are two 8-half runs per lane: pure b128 loads.
__global__ __launch_bounds__(256) void gemm_hprime(const _Float16* __restrict__ hh,
                                                   const _Float16* __restrict__ Wh,
                                                   float* __restrict__ hp,
                                                   _Float16* __restrict__ hpT) {
  const int lane = threadIdx.x & 31;
  const int wave = threadIdx.x >> 5;
  const int tile = blockIdx.x * 8 + wave;          // 8192 tiles total
  const int mt = tile >> 5;                        // 256 row tiles
  const int nt = tile & 31;                        // 32 col tiles
  const int i0 = mt * 16, n0 = nt * 16;
  const int r    = lane & 15;
  const int hi8  = (lane < 16) ? 0 : 8;
  const int hi16 = (lane < 16) ? 0 : 16;

  const _Float16* arow = hh + (size_t)(i0 + r) * IN_DIM + hi8;
  const _Float16* brow = Wh + (size_t)(n0 + r) * IN_DIM + hi8;    // B[k][n]=W[n][k]

  v8f c = {};
  #pragma unroll 4
  for (int kt = 0; kt < IN_DIM / 32; ++kt) {
    const _Float16* ap = arow + kt * 32;
    v8h a0 = *(const v8h*)(ap);
    v8h a1 = *(const v8h*)(ap + 16);
    v16h a = __builtin_shufflevector(a0, a1, 0,1,2,3,4,5,6,7,8,9,10,11,12,13,14,15);
    const _Float16* bp = brow + kt * 32;
    v8h b0 = *(const v8h*)(bp);
    v8h b1 = *(const v8h*)(bp + 16);
    v16h b = __builtin_shufflevector(b0, b1, 0,1,2,3,4,5,6,7,8,9,10,11,12,13,14,15);
    c = __builtin_amdgcn_wmma_f32_16x16x32_f16(false, a, false, b, (short)0, c,
                                               false, false);
  }

  const int col = n0 + r;
  #pragma unroll
  for (int t = 0; t < 8; ++t) {
    float v = c[t];
    int row = i0 + hi8 + t;
    hp[(size_t)row * OUT_DIM + col]  = v;
    hpT[(size_t)col * N_NODES + row] = (_Float16)v;   // contiguous b128 store
  }
}

// --------------------- per-(node,head) logits, pre-scaled into exp2 domain
__global__ __launch_bounds__(256) void attn_logits(const float* __restrict__ hp,
                                                   const float* __restrict__ a_src,
                                                   const float* __restrict__ a_dst,
                                                   float* __restrict__ sbuf,
                                                   float* __restrict__ dbuf,
                                                   float* __restrict__ dbufH) {
  int idx = blockIdx.x * blockDim.x + threadIdx.x;  // idx = i*4 + h
  int i = idx >> 2, hd = idx & 3;
  const float* hrow = hp + (size_t)i * OUT_DIM + hd * HEAD_DIM;
  const float* as = a_src + hd * HEAD_DIM;
  const float* ad = a_dst + hd * HEAD_DIM;
  float ss = 0.f, sd = 0.f;
  #pragma unroll 8
  for (int d = 0; d < HEAD_DIM; ++d) {
    float v = hrow[d];
    ss += v * as[d];
    sd += v * ad[d];
  }
  sbuf[idx] = ss * LOG2E;                       // log2-domain source logit
  sd *= LOG2E;
  dbuf[idx] = sd;                               // j-major (float4 per node)
  dbufH[hd * N_NODES + i] = sd;                 // head-major (contiguous in j)
}

// ------- two-phase masked softmax stats: m2[i,h] (log2 domain), 1/l[i,h]
// Phase 1 uses monotonicity: max_nbr leaky(s+d) = leaky(s + max_nbr d),
// so the max pass is pure fmax (no exp, no serial dependency).
__global__ __launch_bounds__(128) void softmax_stats(const int* __restrict__ adj,
                                                     const float* __restrict__ sbuf,
                                                     const float* __restrict__ dbuf,
                                                     float* __restrict__ mbuf,
                                                     float* __restrict__ linv) {
  const int lane = threadIdx.x & 31;
  const int wave = threadIdx.x >> 5;
  const int i = blockIdx.x * 4 + wave;
  const int* arow = adj + (size_t)i * N_NODES;

  float s2[4], md[4];
  int any = 0;
  #pragma unroll
  for (int h = 0; h < 4; ++h) { s2[h] = sbuf[i * 4 + h]; md[h] = -3.0e38f; }

  // phase 1: per-head max of d over neighbors
  for (int j = lane; j < N_NODES; j += 32) {
    __builtin_prefetch(arow + j + 256, 0, 0);
    int av = arow[j];
    const v4f dv = *(const v4f*)(dbuf + j * 4);
    int nb = (av > 0);
    any |= nb;
    #pragma unroll
    for (int h = 0; h < 4; ++h) md[h] = nb ? fmaxf(md[h], dv[h]) : md[h];
  }
  #pragma unroll
  for (int off = 16; off >= 1; off >>= 1) {
    any |= __shfl_xor(any, off);
    #pragma unroll
    for (int h = 0; h < 4; ++h) md[h] = fmaxf(md[h], __shfl_xor(md[h], off));
  }
  float m2[4];
  #pragma unroll
  for (int h = 0; h < 4; ++h) {
    float x = s2[h] + md[h];
    x = fmaxf(x, ALPHA * x);                     // leaky (monotone)
    m2[h] = any ? x : NEG_INF;                   // isolated node: uniform softmax
  }

  // phase 2: one exp2 per (j,h); adj row is L0/L2-hot from phase 1
  float l[4] = {0.f, 0.f, 0.f, 0.f};
  for (int j = lane; j < N_NODES; j += 32) {
    int av = arow[j];
    const v4f dv = *(const v4f*)(dbuf + j * 4);
    #pragma unroll
    for (int h = 0; h < 4; ++h) {
      float x = s2[h] + dv[h];
      x = fmaxf(x, ALPHA * x);
      x = (av > 0) ? x : NEG_INF;
      l[h] += exp2f(x - m2[h]);
    }
  }
  #pragma unroll
  for (int off = 16; off >= 1; off >>= 1) {
    #pragma unroll
    for (int h = 0; h < 4; ++h) l[h] += __shfl_xor(l[h], off);
  }
  if (lane == 0) {
    #pragma unroll
    for (int h = 0; h < 4; ++h) {
      mbuf[i * 4 + h] = m2[h];
      linv[i * 4 + h] = 1.0f / l[h];
    }
  }
}

// ----------- out[i, hd*128+n] = ELU( sum_j P[i,j] * hp[j, hd*128+n] )  (WMMA)
// 8-wave blocks: 4 row-tiles x 2-way j-split. Each wave builds its exact
// A-fragment slots of P in registers (one exp2 + fmax + cndmask per element),
// B = hpT (K-major, two b128 loads per fragment). Partial sums combined in LDS.
__global__ __launch_bounds__(256) void attn_gemm(const int* __restrict__ adj,
                                                 const float* __restrict__ sbuf,
                                                 const float* __restrict__ dbufH,
                                                 const float* __restrict__ mbuf,
                                                 const float* __restrict__ linv,
                                                 const _Float16* __restrict__ hpT,
                                                 float* __restrict__ out) {
  __shared__ float red[4 * 16 * HEAD_DIM];       // 32 KB of the 320 KB LDS
  const int lane  = threadIdx.x & 31;
  const int wave  = threadIdx.x >> 5;            // 0..7
  const int tt    = wave & 3;                    // row tile within block
  const int jhalf = wave >> 2;                   // j-split
  const int it = blockIdx.x * 4 + tt;            // 256 row tiles
  const int hd = blockIdx.y;                     // 4 heads
  const int i0 = it * 16;
  const int r    = lane & 15;
  const int hi8  = (lane < 16) ? 0 : 8;
  const int hi16 = (lane < 16) ? 0 : 16;
  const int i = i0 + r;                          // row this lane owns

  const float s2_r = sbuf[i * 4 + hd];
  const float m2_r = mbuf[i * 4 + hd];
  const float li_r = linv[i * 4 + hd];
  const int* arow = adj + (size_t)i * N_NODES;
  const float* drow = dbufH + (size_t)hd * N_NODES;
  const _Float16* bbase = hpT + (size_t)(hd * HEAD_DIM) * N_NODES;

  v8f acc[8] = {};                               // 16x128 f32 accumulator

  const int jbeg = jhalf * (N_NODES / 2);
  for (int jt = 0; jt < N_NODES / 64; ++jt) {
    const int j0 = jbeg + jt * 32;
    __builtin_prefetch(arow + j0 + 128, 0, 0);

    // lane's 16 slots are two contiguous runs of 8: [j0+hi8, +8) and [+16, +24)
    const int* ab = arow + j0 + hi8;
    v4i a0 = *(const v4i*)(ab);
    v4i a1 = *(const v4i*)(ab + 4);
    v4i a2 = *(const v4i*)(ab + 16);
    v4i a3 = *(const v4i*)(ab + 20);
    const float* db = drow + j0 + hi8;
    v4f d0 = *(const v4f*)(db);
    v4f d1 = *(const v4f*)(db + 4);
    v4f d2 = *(const v4f*)(db + 16);
    v4f d3 = *(const v4f*)(db + 20);
    int   av[16];
    float dd[16];
    #pragma unroll
    for (int q = 0; q < 4; ++q) {
      av[q] = a0[q]; av[4 + q] = a1[q]; av[8 + q] = a2[q]; av[12 + q] = a3[q];
      dd[q] = d0[q]; dd[4 + q] = d1[q]; dd[8 + q] = d2[q]; dd[12 + q] = d3[q];
    }

    // A fragment: P = exp2(mask(leaky2(s2+d2)) - m2) * (1/l)
    v16h p;
    #pragma unroll
    for (int e = 0; e < 16; ++e) {
      float x = s2_r + dd[e];
      x = fmaxf(x, ALPHA * x);
      x = (av[e] > 0) ? x : NEG_INF;
      p[e] = (_Float16)(exp2f(x - m2_r) * li_r);
    }

    #pragma unroll
    for (int nt = 0; nt < 8; ++nt) {
      const _Float16* bp = bbase + (size_t)(nt * 16 + r) * N_NODES + j0 + hi8;
      v8h b0 = *(const v8h*)(bp);
      v8h b1 = *(const v8h*)(bp + 16);
      v16h b = __builtin_shufflevector(b0, b1, 0,1,2,3,4,5,6,7,8,9,10,11,12,13,14,15);
      acc[nt] = __builtin_amdgcn_wmma_f32_16x16x32_f16(false, p, false, b, (short)0,
                                                       acc[nt], false, false);
    }
  }

  // combine the two j-halves through LDS, then ELU + store from the jhalf==0 wave
  float* myred = red + tt * (16 * HEAD_DIM);
  if (jhalf == 1) {
    #pragma unroll
    for (int nt = 0; nt < 8; ++nt)
      #pragma unroll
      for (int t = 0; t < 8; ++t)
        myred[(hi8 + t) * HEAD_DIM + nt * 16 + r] = acc[nt][t];
  }
  __syncthreads();
  if (jhalf == 0) {
    #pragma unroll
    for (int nt = 0; nt < 8; ++nt) {
      int col = hd * HEAD_DIM + nt * 16 + r;
      #pragma unroll
      for (int t = 0; t < 8; ++t) {
        float v = acc[nt][t] + myred[(hi8 + t) * HEAD_DIM + nt * 16 + r];
        v = (v > 0.f) ? v : (__expf(v) - 1.0f);  // ELU
        out[(size_t)(i0 + hi8 + t) * OUT_DIM + col] = v;
      }
    }
  }
}

// ---------------------------------------------------------------------------
extern "C" void kernel_launch(void* const* d_in, const int* in_sizes, int n_in,
                              void* d_out, int out_size, void* d_ws, size_t ws_size,
                              hipStream_t stream) {
  const float* h     = (const float*)d_in[0];   // (4096,512)
  const int*   adj   = (const int*)d_in[1];     // (4096,4096)
  const float* W     = (const float*)d_in[2];   // (512,512)
  const float* a_src = (const float*)d_in[3];   // (4,128)
  const float* a_dst = (const float*)d_in[4];   // (4,128)
  float* out = (float*)d_out;                   // (4096,512)

  char* ws = (char*)d_ws;
  _Float16* hh    = (_Float16*)(ws + 0);            // 4 MB
  _Float16* Wh    = (_Float16*)(ws + 4194304);      // 0.5 MB
  float*    hp    = (float*)   (ws + 4718592);      // 8 MB
  _Float16* hpT   = (_Float16*)(ws + 13107200);     // 4 MB (K-major f16)
  float*    sbuf  = (float*)   (ws + 17301504);     // 64 KB
  float*    dbuf  = (float*)   (ws + 17367040);     // 64 KB (j-major)
  float*    dbufH = (float*)   (ws + 17432576);     // 64 KB (head-major)
  float*    mbuf  = (float*)   (ws + 17498112);     // 64 KB
  float*    livb  = (float*)   (ws + 17563648);     // 64 KB

  cvt_f32_f16<<<(N_NODES * IN_DIM + 255) / 256, 256, 0, stream>>>(h, hh,
                                                                  N_NODES * IN_DIM);
  cvt_f32_f16<<<(OUT_DIM * IN_DIM + 255) / 256, 256, 0, stream>>>(W, Wh,
                                                                  OUT_DIM * IN_DIM);
  gemm_hprime<<<1024, 256, 0, stream>>>(hh, Wh, hp, hpT);
  attn_logits<<<(N_NODES * NHEADS) / 256, 256, 0, stream>>>(hp, a_src, a_dst,
                                                            sbuf, dbuf, dbufH);
  softmax_stats<<<N_NODES / 4, 128, 0, stream>>>(adj, sbuf, dbuf, mbuf, livb);
  attn_gemm<<<dim3(N_NODES / 64, NHEADS), 256, 0, stream>>>(adj, sbuf, dbufH, mbuf,
                                                            livb, hpT, out);
}
